// CW_NER_36919538876499
// MI455X (gfx1250) — matrix operand order repaired and checked by reading, compiled, not run on hardware
//
#include <hip/hip_runtime.h>

// ---------------- problem constants ----------------
#define BB 64
#define SS 200
#define GG 8
#define DC 128
#define DG 128
#define DIN 256
#define HH 512
#define G4 2048          // 4*H
#define NT 22            // N_LABEL + 2
#define START_TAG 20
#define STOP_TAG 21
#define REC_BLOCKS 128   // persistent recurrence grid (2 dirs x 4 mtiles x 16 hpairs)

typedef __attribute__((ext_vector_type(16))) _Float16 v16h;
typedef __attribute__((ext_vector_type(8)))  _Float16 v8h;
typedef __attribute__((ext_vector_type(8)))  float    v8f;

// ---- A-fragment loader: 16-bit A 16x32 layout (ISA 7.12.2) ----
// lane<16 : row M=lane,    elems 0..7 = K kbase+0..7,  elems 8..15 = K kbase+16..23
// lane>=16: row M=lane-16, elems 0..7 = K kbase+8..15, elems 8..15 = K kbase+24..31
__device__ __forceinline__ v16h ner_load_a(const _Float16* __restrict__ A, int lda,
                                           int row0, int kbase, int lane) {
    int m    = lane & 15;
    int koff = (lane >> 4) << 3;          // 0 or 8
    const _Float16* p = A + (size_t)(row0 + m) * lda + kbase + koff;
    union { v16h v; v8h h[2]; } u;
    u.h[0] = *(const v8h*)(p);
    u.h[1] = *(const v8h*)(p + 16);
    return u.v;
}

// ---- weight swizzle: w[N=2048][K] fp32 row-major -> B-fragment-major f16 ----
// out[ktile][ntile][lane][i] = w[ ntile*16 + (lane&15) ][ ktile*32 + (lane>>4)*8 + (i&7) + (i>>3)*16 ]
__global__ void ner_swizzle_w(const float* __restrict__ w, _Float16* __restrict__ out, int K) {
    size_t gid = (size_t)blockIdx.x * 256 + threadIdx.x;
    size_t total = (size_t)G4 * K;
    if (gid >= total) return;
    size_t idx = gid;
    int i    = idx & 15; idx >>= 4;
    int lane = idx & 31; idx >>= 5;
    int ntile = (int)(idx & 127); idx >>= 7;   // 2048/16 = 128 N-tiles
    int ktile = (int)idx;
    int n = ntile * 16 + (lane & 15);
    int k = ktile * 32 + (lane >> 4) * 8 + (i & 7) + (i >> 3) * 16;
    out[gid] = (_Float16)w[(size_t)n * K + k];
}

// ---- embeddings + masked gaz mean + rev_pad scatter -> f16 LSTM inputs ----
__global__ void ner_prep(const int* __restrict__ char_inputs, const int* __restrict__ lengths,
                         const int* __restrict__ gaz_ids, const int* __restrict__ gaz_mask,
                         const int* __restrict__ rev_gaz_ids, const int* __restrict__ rev_gaz_mask,
                         const float* __restrict__ char_table, const float* __restrict__ gaz_table,
                         _Float16* __restrict__ fwdin, _Float16* __restrict__ bwdin) {
    int bs = blockIdx.x;              // b*SS + s
    int b  = bs / SS, s = bs % SS;
    int d  = threadIdx.x;             // 0..127
    int len  = lengths[b];
    int rpos = (s < len) ? (len - 1 - s) : s;
    size_t fo = (size_t)bs * DIN;
    size_t bo = ((size_t)b * SS + rpos) * DIN;

    float ce = char_table[(size_t)char_inputs[bs] * DC + d];
    fwdin[fo + d] = (_Float16)ce;
    bwdin[bo + d] = (_Float16)ce;

    float sum_f = 0.f, cnt_f = 0.f, sum_b = 0.f, cnt_b = 0.f;
    for (int g = 0; g < GG; ++g) {
        int mf = gaz_mask[bs * GG + g];
        if (mf) { sum_f += gaz_table[(size_t)gaz_ids[bs * GG + g] * DG + d]; cnt_f += 1.f; }
        int mb = rev_gaz_mask[bs * GG + g];
        if (mb) { sum_b += gaz_table[(size_t)rev_gaz_ids[bs * GG + g] * DG + d]; cnt_b += 1.f; }
    }
    fwdin[fo + DC + d] = (_Float16)(sum_f / fmaxf(cnt_f, 1.f));
    bwdin[bo + DC + d] = (_Float16)(sum_b / fmaxf(cnt_b, 1.f));
}

// ---- big input GEMM: X = in[12800x256] @ w_ih^T + b -> f16 [12800x2048] ----
// grid (800, 8, 2dirs), block 256 (8 waves; wave = N-subtile)
__global__ void ner_input_gemm(const _Float16* __restrict__ Af, const _Float16* __restrict__ Ab,
                               const _Float16* __restrict__ Bf, const _Float16* __restrict__ Bb,
                               const float* __restrict__ biasf, const float* __restrict__ biasb,
                               _Float16* __restrict__ Xf, _Float16* __restrict__ Xb) {
    int lane = threadIdx.x & 31;
    int wv   = threadIdx.x >> 5;
    int dir  = blockIdx.z;
    const _Float16* A  = dir ? Ab : Af;
    const _Float16* Bw = dir ? Bb : Bf;
    const float* bias  = dir ? biasb : biasf;
    _Float16* X        = dir ? Xb : Xf;
    int mtile = blockIdx.x;                 // 0..799
    int ntile = blockIdx.y * 8 + wv;        // 0..63

    v8f acc = {};
    for (int kt = 0; kt < DIN / 32; ++kt) {
        v16h a = ner_load_a(A, DIN, mtile * 16, kt * 32, lane);
        v16h b = *(const v16h*)(Bw + (((size_t)kt * 128 + ntile) * 512) + lane * 16);
        acc = __builtin_amdgcn_wmma_f32_16x16x32_f16(false, a, false, b, (short)0, acc, false, false);
    }
    int mb = (lane >> 4) * 8;
    int n  = ntile * 16 + (lane & 15);
    float bv = bias[n];
    for (int r = 0; r < 8; ++r) {
        int row = mtile * 16 + mb + r;
        X[(size_t)row * G4 + n] = (_Float16)(acc[r] + bv);
    }
}

// ---- barrier state init (ws is poisoned; reset every call) ----
__global__ void ner_init_barrier(int* __restrict__ bar_cnt, int* __restrict__ bar_gen) {
    *bar_cnt = 0;
    *bar_gen = 0;
}

// ---- persistent bidirectional LSTM: all 200 steps in one launch ----
// grid REC_BLOCKS blocks of 256 threads; software grid barrier between steps.
// block: dir = bid>>6; mtile = bid&3; hpair = (bid&63)>>2
// wave w (0..7): gate = w>>1, hsub = w&1 -> one 16x16 gate tile per wave per step.
__global__ void ner_lstm_all(const _Float16* __restrict__ Xf, const _Float16* __restrict__ Xb,
                             const _Float16* __restrict__ Wf, const _Float16* __restrict__ Wb,
                             _Float16* __restrict__ hf_all, _Float16* __restrict__ hb_all,
                             float* __restrict__ cf, float* __restrict__ cb,
                             int* __restrict__ bar_cnt, int* __restrict__ bar_gen) {
    __shared__ float g_s[8][16][16];
    int bid   = blockIdx.x;
    int dir   = bid >> 6;
    int rem   = bid & 63;
    int mtile = rem & 3;                 // batch/16
    int hpair = rem >> 2;                // 0..15 (pair of hidden tiles)
    int lane  = threadIdx.x & 31;
    int wv    = threadIdx.x >> 5;        // 0..7
    int gate  = wv >> 1;                 // 0..3 (i,f,g,o)
    int hsub  = wv & 1;
    int htile = hpair * 2 + hsub;        // 0..31
    int ntile = gate * 32 + htile;       // 0..127 within 2048 gate cols

    const _Float16* X = dir ? Xb : Xf;
    const _Float16* W = dir ? Wb : Wf;
    _Float16* h_all   = dir ? hb_all : hf_all;
    float* cst        = dir ? cb : cf;

    for (int t = 0; t < SS; ++t) {
        // gates_tile = h_{t-1} @ w_hh^T  (WMMA, K=512)
        v8f acc = {};
        if (t > 0) {
            const _Float16* hprev = h_all + (size_t)(t - 1) * BB * HH;
            for (int kt = 0; kt < HH / 32; ++kt) {
                v16h a = ner_load_a(hprev, HH, mtile * 16, kt * 32, lane);
                v16h b = *(const v16h*)(W + (((size_t)kt * 128 + ntile) * 512) + lane * 16);
                acc = __builtin_amdgcn_wmma_f32_16x16x32_f16(false, a, false, b, (short)0, acc,
                                                             false, false);
            }
        }
        int mb = (lane >> 4) * 8;
        int nn = lane & 15;
        for (int r = 0; r < 8; ++r) g_s[wv][mb + r][nn] = acc[r];
        __syncthreads();

        // LSTM cell epilogue: 512 (m,col) elements over 256 threads
        for (int E = threadIdx.x; E < 512; E += 256) {
            int hs = E >> 8, m = (E >> 4) & 15, n = E & 15;
            int row = mtile * 16 + m;                  // batch index
            int col = (hpair * 2 + hs) * 16 + n;       // hidden index
            size_t xb = ((size_t)row * SS + t) * G4;
            float gi = g_s[0 * 2 + hs][m][n] + (float)X[xb + 0 * HH + col];
            float gf = g_s[1 * 2 + hs][m][n] + (float)X[xb + 1 * HH + col];
            float gg = g_s[2 * 2 + hs][m][n] + (float)X[xb + 2 * HH + col];
            float go = g_s[3 * 2 + hs][m][n] + (float)X[xb + 3 * HH + col];
            float cold = (t > 0) ? cst[row * HH + col] : 0.f;
            float i_ = 1.f / (1.f + __expf(-gi));
            float f_ = 1.f / (1.f + __expf(-gf));
            float o_ = 1.f / (1.f + __expf(-go));
            float cn = f_ * cold + i_ * tanhf(gg);
            cst[row * HH + col] = cn;
            h_all[((size_t)t * BB + row) * HH + col] = (_Float16)(o_ * tanhf(cn));
        }

        // ---- software grid barrier (release h_t, acquire for t+1) ----
        __threadfence();                       // per-wave release: flush h stores to L2
        __syncthreads();
        if (threadIdx.x == 0) {
            int g = __hip_atomic_load(bar_gen, __ATOMIC_RELAXED, __HIP_MEMORY_SCOPE_AGENT);
            int arrived = __hip_atomic_fetch_add(bar_cnt, 1, __ATOMIC_ACQ_REL,
                                                 __HIP_MEMORY_SCOPE_AGENT);
            if (arrived == REC_BLOCKS - 1) {
                __hip_atomic_store(bar_cnt, 0, __ATOMIC_RELAXED, __HIP_MEMORY_SCOPE_AGENT);
                __hip_atomic_fetch_add(bar_gen, 1, __ATOMIC_RELEASE, __HIP_MEMORY_SCOPE_AGENT);
            } else {
                while (__hip_atomic_load(bar_gen, __ATOMIC_ACQUIRE,
                                         __HIP_MEMORY_SCOPE_AGENT) == g) {
                    __builtin_amdgcn_s_sleep(1);
                }
            }
        }
        __syncthreads();
        __threadfence();                       // acquire: invalidate WGP caches for h_{t}
    }
}

// ---- emissions: concat(h_f, rev_pad(h_b)) @ W_out + b_out ----
__global__ void ner_emissions(const _Float16* __restrict__ hf, const _Float16* __restrict__ hb,
                              const int* __restrict__ lengths,
                              const float* __restrict__ Wout, const float* __restrict__ bout,
                              float* __restrict__ emis) {
    int gid = blockIdx.x * 256 + threadIdx.x;
    if (gid >= BB * SS * NT) return;
    int tag = gid % NT;
    int bs  = gid / NT;
    int b = bs / SS, s = bs % SS;
    int len = lengths[b];
    int sb  = (s < len) ? (len - 1 - s) : s;
    const _Float16* hfp = hf + ((size_t)s  * BB + b) * HH;
    const _Float16* hbp = hb + ((size_t)sb * BB + b) * HH;
    float acc = bout[tag];
    for (int k8 = 0; k8 < HH / 8; ++k8) {
        v8h hv = *(const v8h*)(hfp + k8 * 8);
        for (int j = 0; j < 8; ++j) acc += (float)hv[j] * Wout[(k8 * 8 + j) * NT + tag];
    }
    for (int k8 = 0; k8 < HH / 8; ++k8) {
        v8h hv = *(const v8h*)(hbp + k8 * 8);
        for (int j = 0; j < 8; ++j) acc += (float)hv[j] * Wout[(HH + k8 * 8 + j) * NT + tag];
    }
    emis[(size_t)bs * NT + tag] = acc;
}

// ---- Viterbi decode: one block per batch element, 32 threads ----
__global__ void ner_viterbi(const float* __restrict__ emis, const int* __restrict__ lengths,
                            const float* __restrict__ trans, int* __restrict__ bp,
                            float* __restrict__ out) {
    __shared__ float part[32];
    __shared__ float tr[NT * NT];
    int b = blockIdx.x;
    int lane = threadIdx.x;
    for (int i = lane; i < NT * NT; i += 32) tr[i] = trans[i];
    int len = lengths[b];
    if (lane < NT) part[lane] = emis[((size_t)b * SS) * NT + lane] + tr[START_TAG * NT + lane];
    __syncthreads();

    for (int t = 1; t < SS; ++t) {
        float newv = 0.f; int bi = 0;
        if (lane < NT) {
            float e = emis[((size_t)b * SS + t) * NT + lane];
            float best = -3.4e38f;
            for (int i = 0; i < NT; ++i) {
                float c = part[i] + tr[i * NT + lane];
                if (c > best) { best = c; bi = i; }
            }
            newv = best + e;
        }
        __syncthreads();
        if (lane < NT) {
            bool m = (t < len);
            if (m) part[lane] = newv;
            bp[((size_t)b * (SS - 1) + (t - 1)) * NT + lane] = m ? bi : lane;
        }
        __syncthreads();
    }

    if (lane == 0) {
        float best = -3.4e38f; int tag = 0;
        for (int j = 0; j < NT; ++j) {
            float v = part[j] + tr[j * NT + STOP_TAG];
            if (v > best) { best = v; tag = j; }
        }
        out[(size_t)b * SS + (SS - 1)] = ((SS - 1) < len) ? (float)tag : 0.f;
        for (int t = SS - 2; t >= 0; --t) {
            tag = bp[((size_t)b * (SS - 1) + t) * NT + tag];
            out[(size_t)b * SS + t] = (t < len) ? (float)tag : 0.f;
        }
    }
}

extern "C" void kernel_launch(void* const* d_in, const int* in_sizes, int n_in,
                              void* d_out, int out_size, void* d_ws, size_t ws_size,
                              hipStream_t stream) {
    const int*   char_inputs  = (const int*)d_in[0];
    const int*   lengths      = (const int*)d_in[1];
    /* d_in[2] = mask: recomputed from lengths */
    const int*   gaz_ids      = (const int*)d_in[3];
    const int*   gaz_mask     = (const int*)d_in[4];
    const int*   rev_gaz_ids  = (const int*)d_in[5];
    const int*   rev_gaz_mask = (const int*)d_in[6];
    const float* char_table   = (const float*)d_in[7];
    const float* gaz_table    = (const float*)d_in[8];
    const float* w_ih_f       = (const float*)d_in[9];
    const float* w_hh_f       = (const float*)d_in[10];
    const float* b_f          = (const float*)d_in[11];
    const float* w_ih_b       = (const float*)d_in[12];
    const float* w_hh_b       = (const float*)d_in[13];
    const float* b_b          = (const float*)d_in[14];
    const float* W_out        = (const float*)d_in[15];
    const float* b_out        = (const float*)d_in[16];
    const float* transitions  = (const float*)d_in[17];
    float* out = (float*)d_out;

    char* p = (char*)d_ws;
    auto alloc = [&](size_t bytes) { char* r = p; p += (bytes + 255) & ~(size_t)255; return r; };
    _Float16* fwdin   = (_Float16*)alloc((size_t)BB * SS * DIN * 2);
    _Float16* bwdin   = (_Float16*)alloc((size_t)BB * SS * DIN * 2);
    _Float16* wihf_sw = (_Float16*)alloc((size_t)DIN * G4 * 2);
    _Float16* wihb_sw = (_Float16*)alloc((size_t)DIN * G4 * 2);
    _Float16* whhf_sw = (_Float16*)alloc((size_t)HH * G4 * 2);
    _Float16* whhb_sw = (_Float16*)alloc((size_t)HH * G4 * 2);
    _Float16* Xf      = (_Float16*)alloc((size_t)BB * SS * G4 * 2);
    _Float16* Xb      = (_Float16*)alloc((size_t)BB * SS * G4 * 2);
    _Float16* hf_all  = (_Float16*)alloc((size_t)SS * BB * HH * 2);
    _Float16* hb_all  = (_Float16*)alloc((size_t)SS * BB * HH * 2);
    float*    cf      = (float*)alloc((size_t)BB * HH * 4);
    float*    cb      = (float*)alloc((size_t)BB * HH * 4);
    float*    emis    = (float*)alloc((size_t)BB * SS * NT * 4);
    int*      bp      = (int*)alloc((size_t)BB * (SS - 1) * NT * 4);
    int*      bar     = (int*)alloc(256);      // [0]=count, [1]=generation

    // 0) reset grid-barrier state (ws not re-initialized between replays)
    ner_init_barrier<<<1, 1, 0, stream>>>(bar, bar + 1);
    // 1) weight swizzles into WMMA B-fragment order (f16)
    {
        int blk_ih = (G4 * DIN + 255) / 256;
        int blk_hh = (G4 * HH + 255) / 256;
        ner_swizzle_w<<<blk_ih, 256, 0, stream>>>(w_ih_f, wihf_sw, DIN);
        ner_swizzle_w<<<blk_ih, 256, 0, stream>>>(w_ih_b, wihb_sw, DIN);
        ner_swizzle_w<<<blk_hh, 256, 0, stream>>>(w_hh_f, whhf_sw, HH);
        ner_swizzle_w<<<blk_hh, 256, 0, stream>>>(w_hh_b, whhb_sw, HH);
    }
    // 2) embeddings + gaz means + rev_pad scatter
    ner_prep<<<BB * SS, DC, 0, stream>>>(char_inputs, lengths, gaz_ids, gaz_mask,
                                         rev_gaz_ids, rev_gaz_mask, char_table, gaz_table,
                                         fwdin, bwdin);
    // 3) input-to-hidden GEMM, both directions (WMMA)
    ner_input_gemm<<<dim3(BB * SS / 16, 8, 2), 256, 0, stream>>>(
        fwdin, bwdin, wihf_sw, wihb_sw, b_f, b_b, Xf, Xb);
    // 4) all 200 recurrent steps in one persistent kernel (WMMA + grid barrier)
    ner_lstm_all<<<REC_BLOCKS, 256, 0, stream>>>(
        Xf, Xb, whhf_sw, whhb_sw, hf_all, hb_all, cf, cb, bar, bar + 1);
    // 5) projection to tag emissions
    ner_emissions<<<(BB * SS * NT + 255) / 256, 256, 0, stream>>>(
        hf_all, hb_all, lengths, W_out, b_out, emis);
    // 6) Viterbi decode + masked tag output
    ner_viterbi<<<BB, 32, 0, stream>>>(emis, lengths, transitions, bp, out);
    (void)in_sizes; (void)n_in; (void)out_size; (void)ws_size;
}
